// Network_13013750906990
// MI455X (gfx1250) — matrix operand order repaired and checked by reading, not compiled, not run on hardware
//
#include <hip/hip_runtime.h>
#include <stdint.h>

typedef __attribute__((ext_vector_type(16))) _Float16 v16h;
typedef __attribute__((ext_vector_type(8)))  float    v8f;

namespace {
constexpr int kB = 32, kT = 300, kM = kB * kT;      // 9600 rows (b*T + t)
constexpr int kFin  = 3072;
constexpr int kFhid = 410, kFout = 10;
constexpr int kHP   = 448;            // hidden padded to 28 n-tiles / 14 k-tiles
constexpr int kNP2  = 16;             // output padded 10 -> 16
constexpr int kKT1  = kFin / 32;      // 96  K-steps GEMM1
constexpr int kMT   = kM / 16;        // 600 M-tiles
constexpr int kNT1  = kHP / 16;       // 28  N-tiles GEMM1
constexpr int kKT2  = kHP / 32;       // 14  K-steps GEMM2
constexpr float kTheta = 10.0f;
constexpr float kDecay = 0.9048374180359595f;   // exp(-TS/TAU_SR)

// ---- workspace layout (bytes), everything 256B aligned ----
constexpr size_t offSpk = 0;
constexpr size_t szSpk  = (size_t)kMT * kKT1 * 512 * 2;   // 58,982,400 (A-frags L1)
constexpr size_t offW1  = offSpk + szSpk;
constexpr size_t szW1   = (size_t)kKT1 * kNT1 * 512 * 2;  //  2,752,512 (B-frags L1)
constexpr size_t offA1  = offW1 + szW1;
constexpr size_t szA1   = (size_t)kM * kHP * 4;           // 17,203,200
constexpr size_t offS1  = offA1 + szA1;
constexpr size_t szS1   = (size_t)kMT * kKT2 * 512 * 2;   //  8,601,600 (A-frags L2)
constexpr size_t offW2  = offS1 + szS1;
constexpr size_t szW2   = (size_t)kKT2 * 512 * 2;         //     14,336 (B-frags L2)
constexpr size_t offA2  = offW2 + szW2;                   // a2: 9600*16*4
} // namespace

// -------- helpers --------

// stateless uniform [0,1): murmur3-style fmix of (row, feature)
__device__ __forceinline__ float hash_u01(uint32_t m, uint32_t f) {
    uint32_t x = m * 0x9E3779B9u ^ (f * 0x85EBCA77u) ^ 0x27D4EB2Fu;
    x ^= x >> 16; x *= 0x7FEB352Du;
    x ^= x >> 15; x *= 0x846CA68Bu;
    x ^= x >> 16;
    return (float)(x >> 8) * (1.0f / 16777216.0f);
}

// A-fragment (16x32 f16): lane<16 holds row=lane,    K {0..7,16..23};
//                         lane>=16 holds row=lane-16, K {8..15,24..31}
__device__ __forceinline__ int a_klocal(int lane, int e) {
    return e + (e & 8) + ((lane >> 4) << 3);
}
// B-fragment (32x16 f16): lane<16 -> col=lane, K {0..7,16..23} ; lane>=16 -> col=lane-16, K {8..15,24..31}
__device__ __forceinline__ int b_klocal(int lane, int e) {
    return e + (e & 8) + ((lane >> 4) << 3);
}

__device__ __forceinline__ void store_tile(float* __restrict__ C, int ldc,
                                           int mBase, int nBase, int lane, v8f c) {
    const int r0 = mBase + ((lane >> 4) << 3);
    const int n  = nBase + (lane & 15);
#pragma unroll
    for (int v = 0; v < 8; ++v) C[(size_t)(r0 + v) * ldc + n] = c[v];
}

// -------- 1) rate encoding, written directly in WMMA A-fragment layout --------
__global__ void encode_kernel(const float* __restrict__ inp, _Float16* __restrict__ spk) {
    int idx  = blockIdx.x * blockDim.x + threadIdx.x;  // one f16 element each
    int e    = idx & 15;
    int lane = (idx >> 4) & 31;
    int blk  = idx >> 9;            // [mt][kt] block
    int kt   = blk % kKT1;
    int mt   = blk / kKT1;
    int m    = mt * 16 + (lane & 15);
    int f    = kt * 32 + a_klocal(lane, e);
    int b    = m / kT;
    float prob = inp[(size_t)b * kFin + f];
    float u    = hash_u01((uint32_t)m, (uint32_t)f);
    spk[idx] = (u < prob) ? (_Float16)1.0f : (_Float16)0.0f;
}

// -------- 2) pack W1^T into B-fragment layout ([kt][nt] blocks), zero-padded --------
__global__ void pack_w1_kernel(const float* __restrict__ W1, _Float16* __restrict__ w1f) {
    int idx  = blockIdx.x * blockDim.x + threadIdx.x;
    int e    = idx & 15;
    int lane = (idx >> 4) & 31;
    int blk  = idx >> 9;
    int nt   = blk % kNT1;
    int kt   = blk / kNT1;
    int k    = kt * 32 + b_klocal(lane, e);
    int n    = nt * 16 + (lane & 15);
    w1f[idx] = (n < kFhid) ? (_Float16)W1[(size_t)n * kFin + k] : (_Float16)0.0f;
}

__global__ void pack_w2_kernel(const float* __restrict__ W2, _Float16* __restrict__ w2f) {
    int idx  = blockIdx.x * blockDim.x + threadIdx.x;
    int e    = idx & 15;
    int lane = (idx >> 4) & 31;
    int kt   = idx >> 9;
    int k    = kt * 32 + b_klocal(lane, e);
    int n    = lane & 15;
    w2f[idx] = (n < kFout && k < kFhid) ? (_Float16)W2[(size_t)n * kFhid + k]
                                        : (_Float16)0.0f;
}

// -------- 3) GEMM1: [9600 x 3072] x [3072 x 448] -> a1 f32 --------
// block = 8 waves; block tile 128x64; wave tile 32x32 (2x2 WMMA accumulators)
__global__ void __launch_bounds__(256) gemm1_kernel(const _Float16* __restrict__ A,
                                                    const _Float16* __restrict__ Bf,
                                                    float* __restrict__ C) {
    const int lane = threadIdx.x & 31;
    const int w    = threadIdx.x >> 5;
    const int wr   = w & 3, wc = w >> 2;
    const int mt0  = blockIdx.x * 8 + wr * 2;
    const int nt0  = blockIdx.y * 4 + wc * 2;

    v8f acc00 = {}, acc01 = {}, acc10 = {}, acc11 = {};

    const _Float16* aP0 = A + ((size_t)(mt0 + 0) * kKT1) * 512 + lane * 16;
    const _Float16* aP1 = A + ((size_t)(mt0 + 1) * kKT1) * 512 + lane * 16;
    const _Float16* bP0 = Bf + ((size_t)nt0 + 0) * 512 + lane * 16;
    const _Float16* bP1 = Bf + ((size_t)nt0 + 1) * 512 + lane * 16;

    for (int kt = 0; kt < kKT1; ++kt) {
        v16h a0 = *(const v16h*)(aP0 + (size_t)kt * 512);
        v16h a1 = *(const v16h*)(aP1 + (size_t)kt * 512);
        v16h b0 = *(const v16h*)(bP0 + (size_t)kt * kNT1 * 512);
        v16h b1 = *(const v16h*)(bP1 + (size_t)kt * kNT1 * 512);
        if (kt + 1 < kKT1) {   // uniform branch; WGP-scope prefetch for next K-step
            __builtin_prefetch(aP0 + (size_t)(kt + 1) * 512, 0, 3);
            __builtin_prefetch(aP1 + (size_t)(kt + 1) * 512, 0, 3);
            __builtin_prefetch(bP0 + (size_t)(kt + 1) * kNT1 * 512, 0, 3);
            __builtin_prefetch(bP1 + (size_t)(kt + 1) * kNT1 * 512, 0, 3);
        }
        acc00 = __builtin_amdgcn_wmma_f32_16x16x32_f16(false, a0, false, b0, (short)0, acc00, false, false);
        acc01 = __builtin_amdgcn_wmma_f32_16x16x32_f16(false, a0, false, b1, (short)0, acc01, false, false);
        acc10 = __builtin_amdgcn_wmma_f32_16x16x32_f16(false, a1, false, b0, (short)0, acc10, false, false);
        acc11 = __builtin_amdgcn_wmma_f32_16x16x32_f16(false, a1, false, b1, (short)0, acc11, false, false);
    }
    store_tile(C, kHP, (mt0 + 0) * 16, (nt0 + 0) * 16, lane, acc00);
    store_tile(C, kHP, (mt0 + 0) * 16, (nt0 + 1) * 16, lane, acc01);
    store_tile(C, kHP, (mt0 + 1) * 16, (nt0 + 0) * 16, lane, acc10);
    store_tile(C, kHP, (mt0 + 1) * 16, (nt0 + 1) * 16, lane, acc11);
}

// -------- 4) PSP scan + threshold (layer 1); emits spikes in A-fragment layout --------
__global__ void psp_spike1_kernel(const float* __restrict__ A1, _Float16* __restrict__ S1) {
    int id = blockIdx.x * blockDim.x + threadIdx.x;     // one (b, o) per thread
    if (id >= kB * kHP) return;
    int o  = id % kHP;
    int b  = id / kHP;
    int kt = o >> 5;
    int kl = o & 31;
    int hi = (kl >> 3) & 1;                 // lane half from K-local
    int e  = (kl & 7) + ((kl >> 4) << 3);   // fragment element index
    float u = 0.0f;
    for (int t = 0; t < kT; ++t) {
        int m = b * kT + t;
        u = kDecay * u + A1[(size_t)m * kHP + o];
        _Float16 s = (u >= kTheta) ? (_Float16)1.0f : (_Float16)0.0f;
        int lane = (m & 15) + (hi << 4);
        S1[((size_t)(m >> 4) * kKT2 + kt) * 512 + lane * 16 + e] = s;
    }
}

// -------- 5) GEMM2: [9600 x 448] x [448 x 16] -> a2 f32; one 16x16 tile per wave --------
__global__ void __launch_bounds__(256) gemm2_kernel(const _Float16* __restrict__ S1,
                                                    const _Float16* __restrict__ W2f,
                                                    float* __restrict__ A2) {
    const int lane = threadIdx.x & 31;
    const int w    = threadIdx.x >> 5;
    const int mt   = blockIdx.x * 8 + w;
    v8f acc = {};
    for (int kt = 0; kt < kKT2; ++kt) {
        v16h a = *(const v16h*)(S1 + ((size_t)mt * kKT2 + kt) * 512 + lane * 16);
        v16h b = *(const v16h*)(W2f + (size_t)kt * 512 + lane * 16);
        acc = __builtin_amdgcn_wmma_f32_16x16x32_f16(false, a, false, b, (short)0, acc, false, false);
    }
    store_tile(A2, kNP2, mt * 16, 0, lane, acc);
}

// -------- 6) PSP scan + threshold (layer 2) -> out [B, 10, T] f32 --------
__global__ void psp_spike2_kernel(const float* __restrict__ A2, float* __restrict__ out) {
    int id = blockIdx.x * blockDim.x + threadIdx.x;
    if (id >= kB * kNP2) return;
    int o = id & 15;
    int b = id >> 4;
    if (o >= kFout) return;
    float u = 0.0f;
    for (int t = 0; t < kT; ++t) {
        u = kDecay * u + A2[(size_t)(b * kT + t) * kNP2 + o];
        out[((size_t)b * kFout + o) * kT + t] = (u >= kTheta) ? 1.0f : 0.0f;
    }
}

extern "C" void kernel_launch(void* const* d_in, const int* in_sizes, int n_in,
                              void* d_out, int out_size, void* d_ws, size_t ws_size,
                              hipStream_t stream) {
    (void)in_sizes; (void)n_in; (void)out_size; (void)ws_size;
    const float* inp = (const float*)d_in[0];   // [32,3,32,32]
    const float* W1  = (const float*)d_in[1];   // [410,3072]
    const float* W2  = (const float*)d_in[2];   // [10,410]
    float* out = (float*)d_out;                 // [32,10,300]

    char* ws = (char*)d_ws;
    _Float16* spk = (_Float16*)(ws + offSpk);
    _Float16* w1f = (_Float16*)(ws + offW1);
    float*    a1  = (float*)   (ws + offA1);
    _Float16* s1  = (_Float16*)(ws + offS1);
    _Float16* w2f = (_Float16*)(ws + offW2);
    float*    a2  = (float*)   (ws + offA2);

    encode_kernel <<<kMT * kKT1 * 512 / 256, 256, 0, stream>>>(inp, spk);
    pack_w1_kernel<<<kKT1 * kNT1 * 512 / 256, 256, 0, stream>>>(W1, w1f);
    pack_w2_kernel<<<kKT2 * 512 / 256,        256, 0, stream>>>(W2, w2f);
    gemm1_kernel  <<<dim3(kMT / 8, kNT1 / 4), 256, 0, stream>>>(spk, w1f, a1);
    psp_spike1_kernel<<<(kB * kHP + 255) / 256, 256, 0, stream>>>(a1, s1);
    gemm2_kernel  <<<kMT / 8, 256, 0, stream>>>(s1, w2f, a2);
    psp_spike2_kernel<<<(kB * kNP2 + 255) / 256, 256, 0, stream>>>(a2, out);
}
